// NormalizingFlow_39745627357838
// MI455X (gfx1250) — matrix operand order repaired and checked by reading, not compiled, not run on hardware
//
#include <hip/hip_runtime.h>
#include <hip/hip_bf16.h>
#include <math.h>

#define TT 1024
#define BB 256
#define DD 64
#define EE 512
#define HH 256
#define LL 4
#define DE 576  // D + E

typedef __attribute__((ext_vector_type(16))) __bf16 v16bf;
typedef __attribute__((ext_vector_type(8)))  __bf16 v8bf;
typedef __attribute__((ext_vector_type(8)))  float  v8f;

// Workspace layout (bf16 element offsets)
#define SZ_WINT  (LL * HH * DE)   // W_in transposed per layer: [l][n][k]
#define SZ_WST   (LL * DD * HH)   // W_s transposed per layer:  [l][n][k]
#define SZ_WTT   (LL * DD * HH)   // W_t transposed per layer:  [l][n][k]
#define SZ_WESN  (EE * DE)        // fused [W_inp ; W_res^T] as [e][k]
#define OFF_WINT 0
#define OFF_WST  (OFF_WINT + SZ_WINT)
#define OFF_WTT  (OFF_WST + SZ_WST)
#define OFF_WESN (OFF_WTT + SZ_WTT)
#define TOTAL_W  (OFF_WESN + SZ_WESN)

// Native CDNA5 V_TANH_F32 if available; else branch-free __expf-based tanh.
__device__ __forceinline__ float tanh_fast(float x) {
#if __has_builtin(__builtin_amdgcn_tanhf)
    return __builtin_amdgcn_tanhf(x);
#else
    float e = __expf(-2.f * x);          // v_exp_f32, branch-free
    return (1.f - e) / (1.f + e);
#endif
}

// ---------------------------------------------------------------------------
// A fragment: 16x32 bf16 A-matrix, contiguous K order (same as fragB).
// lane L (<16): row M=L,    holds K = k0+{0..15}
// lane L (>=16): row M=L-16, holds K = k0+{16..31}
__device__ __forceinline__ v16bf fragA(const __bf16* buf, int stride, int lane, int k0) {
    int r  = lane & 15;
    int hi = (lane >> 4) << 4;                 // 0 or 16
    const __bf16* p = buf + r * stride + k0 + hi;
    v16bf v;
    v = *(const v16bf*)(p);                    // K = k0+hi .. k0+hi+15
    return v;
}

// B fragment: 32x16 bf16 B-matrix; buf is n-major (row = n, K contiguous).
// lane L (<16): col N=L, K = k0+0..15 ; lane L (>=16): col N=L-16, K = k0+16..31
__device__ __forceinline__ v16bf fragB(const __bf16* buf, int stride, int lane, int k0) {
    int n  = lane & 15;
    int kk = k0 + ((lane >> 4) << 4);
    return *(const v16bf*)(buf + n * stride + kk);
}

#define WMMA_BF16(a, b, c) \
    __builtin_amdgcn_wmma_f32_16x16x32_bf16(false, (a), false, (b), (short)0, (c), false, false)

// ---------------------------------------------------------------------------
// Prologue: convert weights to bf16 with B-friendly (n-major) layouts.
extern "C" __global__ void prep_weights(const float* __restrict__ W_in,
                                        const float* __restrict__ W_s,
                                        const float* __restrict__ W_t,
                                        const float* __restrict__ W_res,
                                        const float* __restrict__ W_inp,
                                        __bf16* __restrict__ ws) {
    int i = blockIdx.x * blockDim.x + threadIdx.x;
    if (i < SZ_WINT) {           // [l][n][k] = W_in[l][k][n], (L, D+E, H)
        int l = i / (HH * DE); int rem = i % (HH * DE);
        int n = rem / DE;      int k = rem % DE;
        ws[OFF_WINT + i] = (__bf16)W_in[((size_t)l * DE + k) * HH + n];
        return;
    }
    i -= SZ_WINT;
    if (i < SZ_WST) {            // [l][n][k] = W_s[l][k][n], (L, H, D)
        int l = i / (DD * HH); int rem = i % (DD * HH);
        int n = rem / HH;      int k = rem % HH;
        ws[OFF_WST + i] = (__bf16)W_s[((size_t)l * HH + k) * DD + n];
        return;
    }
    i -= SZ_WST;
    if (i < SZ_WTT) {
        int l = i / (DD * HH); int rem = i % (DD * HH);
        int n = rem / HH;      int k = rem % HH;
        ws[OFF_WTT + i] = (__bf16)W_t[((size_t)l * HH + k) * DD + n];
        return;
    }
    i -= SZ_WTT;
    if (i < SZ_WESN) {           // [e][k]: k<64 -> W_inp[k][e]; else W_res[e][k-64]
        int n = i / DE; int k = i % DE;
        float v = (k < DD) ? W_inp[(size_t)k * EE + n]
                           : W_res[(size_t)n * EE + (k - DD)];
        ws[OFF_WESN + i] = (__bf16)v;
        return;
    }
}

// ---------------------------------------------------------------------------
// Main kernel: 16 blocks x 256 threads (8 wave32). Block b owns batch rows
// [16b, 16b+16) and iterates the full T recurrence.
// waves_per_eu(4) caps VGPRs at 256/wave: no vgpr-MSB juggling, no spills.
extern "C" __global__ __launch_bounds__(256)
__attribute__((amdgpu_waves_per_eu(4)))
void flow_esn_kernel(const float* __restrict__ x,
                     const int*   __restrict__ seq_lengths,
                     const float* __restrict__ b_mask,
                     const float* __restrict__ b_in,
                     const float* __restrict__ b_s,
                     const float* __restrict__ b_t,
                     const float* __restrict__ rescale_w,
                     const __bf16* __restrict__ ws,
                     float* __restrict__ out) {
    __shared__ __align__(32) float  zbuf[16 * DD];      // current z (f32)
    __shared__ __align__(32) __bf16 actbuf[16 * DE];    // [masked z | h] bf16
    __shared__ __align__(32) __bf16 xfbuf[16 * DD];     // raw frame x bf16
    __shared__ __align__(32) __bf16 hidbuf[16 * HH];    // hidden bf16
    __shared__ __align__(32) float  slopebuf[16 * DD];
    __shared__ __align__(32) float  intbuf[16 * DD];
    __shared__ float llf[16];
    __shared__ float llacc[16];
    __shared__ float bmk[DD];
    __shared__ float rsc[DD];
    __shared__ int   slen[16];

    const int tid  = threadIdx.x;
    const int lane = tid & 31;
    const int w    = tid >> 5;
    const int m0   = blockIdx.x * 16;

    const __bf16* WinT = ws + OFF_WINT;
    const __bf16* WsT  = ws + OFF_WST;
    const __bf16* WtT  = ws + OFF_WTT;
    const __bf16* Wesn = ws + OFF_WESN;

    for (int i = tid; i < 16 * DE; i += 256) actbuf[i] = (__bf16)0.f;  // h0 = 0
    if (tid < DD) { bmk[tid] = b_mask[tid]; rsc[tid] = rescale_w[tid]; }
    if (tid < 16) { slen[tid] = seq_lengths[m0 + tid]; llacc[tid] = 0.f; }
    __syncthreads();

    const int r4 = tid >> 4;           // row 0..15 for elementwise passes
    const int c4 = (tid & 15) << 2;    // 4 columns per thread

    for (int t = 0; t < TT; ++t) {
        // ---- frame load + prefetch of next frame ----------------------
        {
            const float* xf = x + ((size_t)t * BB + m0) * DD;
            #pragma unroll
            for (int j = 0; j < 4; ++j) {
                float v = xf[r4 * DD + c4 + j];
                zbuf[r4 * DD + c4 + j]  = v;
                xfbuf[r4 * DD + c4 + j] = (__bf16)v;
            }
            if (tid < 16) llf[tid] = 0.f;
            if (t + 1 < TT) {
                // 16x64 f32 frame = 4 KB; 256 threads x 16B apart covers it.
                const float* xn = x + ((size_t)(t + 1) * BB + m0) * DD;
                __builtin_prefetch(xn + tid * 4, 0, 1);   // global_prefetch_b8
            }
        }
        __syncthreads();

        // ---- flow layers (reversed), two masked passes each -----------
        #pragma unroll 1
        for (int l = LL - 1; l >= 0; --l) {
            #pragma unroll 1
            for (int pass = 0; pass < 2; ++pass) {
                // masked z -> actbuf[:, 0:64]
                #pragma unroll
                for (int j = 0; j < 4; ++j) {
                    int d = c4 + j;
                    float m = (pass == 0) ? (1.f - bmk[d]) : bmk[d];
                    actbuf[r4 * DE + d] = (__bf16)(m * zbuf[r4 * DD + d]);
                }
                __syncthreads();

                // GEMM1: hid = tanh(act(16x576) @ Win(576x256) + b_in)
                // Both N-tiles of this wave share the A fragment per K step.
                {
                    const __bf16* Wl0 = WinT + (size_t)l * HH * DE
                                             + (size_t)(2 * w) * 16 * DE;
                    const __bf16* Wl1 = Wl0 + (size_t)16 * DE;
                    v8f acc0 = {}, acc1 = {};
                    #pragma unroll 2
                    for (int k0 = 0; k0 < DE; k0 += 32) {
                        v16bf a = fragA(actbuf, DE, lane, k0);
                        acc0 = WMMA_BF16(a, fragB(Wl0, DE, lane, k0), acc0);
                        acc1 = WMMA_BF16(a, fragB(Wl1, DE, lane, k0), acc1);
                    }
                    const float* bi = b_in + l * HH;
                    int coln  = lane & 15;
                    int rbase = (lane >> 4) << 3;
                    int col0  = (2 * w) * 16 + coln;
                    int col1  = col0 + 16;
                    float bv0 = bi[col0], bv1 = bi[col1];
                    #pragma unroll
                    for (int i = 0; i < 8; ++i) {
                        hidbuf[(rbase + i) * HH + col0] = (__bf16)tanh_fast(acc0[i] + bv0);
                        hidbuf[(rbase + i) * HH + col1] = (__bf16)tanh_fast(acc1[i] + bv1);
                    }
                }
                __syncthreads();

                // GEMM2: slope (waves 0-3) / intercept (waves 4-7)
                {
                    int nt = w & 3;
                    const __bf16* Wb = ((w < 4) ? WsT : WtT)
                                     + (size_t)l * DD * HH + (size_t)nt * 16 * HH;
                    v8f acc = {};
                    #pragma unroll 2
                    for (int k0 = 0; k0 < HH; k0 += 32) {
                        v16bf a = fragA(hidbuf, HH, lane, k0);
                        v16bf b = fragB(Wb, HH, lane, k0);
                        acc = WMMA_BF16(a, b, acc);
                    }
                    int d     = nt * 16 + (lane & 15);
                    int rbase = (lane >> 4) << 3;
                    if (w < 4) {
                        float bv = b_s[l * DD + d], rw = rsc[d];
                        #pragma unroll
                        for (int i = 0; i < 8; ++i)
                            slopebuf[(rbase + i) * DD + d] = tanh_fast(acc[i] + bv) * rw;
                    } else {
                        float bv = b_t[l * DD + d];
                        #pragma unroll
                        for (int i = 0; i < 8; ++i)
                            intbuf[(rbase + i) * DD + d] = acc[i] + bv;
                    }
                }
                __syncthreads();

                // flow update + log-det
                {
                    float part = 0.f;
                    #pragma unroll
                    for (int j = 0; j < 4; ++j) {
                        int d = c4 + j;
                        float m  = (pass == 0) ? (1.f - bmk[d]) : bmk[d];
                        float zv = zbuf[r4 * DD + d];
                        float sl = slopebuf[r4 * DD + d];
                        float ic = intbuf[r4 * DD + d];
                        zbuf[r4 * DD + d] =
                            m * zv + (1.f - m) * ((zv - ic) * __expf(-sl));
                        part += sl * (m - 1.f);
                    }
                    atomicAdd(&llf[r4], part);
                }
                __syncthreads();
            }
        }

        // ---- frame log-likelihood finalize ----------------------------
        {
            float part = 0.f;
            #pragma unroll
            for (int j = 0; j < 4; ++j) {
                float zv = zbuf[r4 * DD + c4 + j];
                part -= 0.5f * zv * zv;
            }
            atomicAdd(&llf[r4], part);
        }
        __syncthreads();
        if (tid < 16) {
            float lf = llf[tid] - 0.5f * (float)DD * 1.8378770664093453f;
            if (t < slen[tid]) llacc[tid] += lf;
        }

        // ---- ESN recurrence: h = tanh([x|h](16x576) @ Wesn(576x512)) --
        {
            v8f acc0 = {}, acc1 = {}, acc2 = {}, acc3 = {};
            const __bf16* Wb0 = Wesn + (size_t)(w * 4 + 0) * 16 * DE;
            const __bf16* Wb1 = Wesn + (size_t)(w * 4 + 1) * 16 * DE;
            const __bf16* Wb2 = Wesn + (size_t)(w * 4 + 2) * 16 * DE;
            const __bf16* Wb3 = Wesn + (size_t)(w * 4 + 3) * 16 * DE;
            #pragma unroll 2
            for (int k0 = 0; k0 < DE; k0 += 32) {
                v16bf a = (k0 < DD) ? fragA(xfbuf, DD, lane, k0)
                                    : fragA(actbuf, DE, lane, k0); // h region
                acc0 = WMMA_BF16(a, fragB(Wb0, DE, lane, k0), acc0);
                acc1 = WMMA_BF16(a, fragB(Wb1, DE, lane, k0), acc1);
                acc2 = WMMA_BF16(a, fragB(Wb2, DE, lane, k0), acc2);
                acc3 = WMMA_BF16(a, fragB(Wb3, DE, lane, k0), acc3);
            }
            __syncthreads();   // all reads of actbuf h done before rewrite
            int rbase = (lane >> 4) << 3;
            int e0 = (w * 4) * 16 + (lane & 15);
            #pragma unroll
            for (int i = 0; i < 8; ++i) {
                int row = (rbase + i) * DE + DD;
                actbuf[row + e0     ] = (__bf16)tanh_fast(acc0[i]);
                actbuf[row + e0 + 16] = (__bf16)tanh_fast(acc1[i]);
                actbuf[row + e0 + 32] = (__bf16)tanh_fast(acc2[i]);
                actbuf[row + e0 + 48] = (__bf16)tanh_fast(acc3[i]);
            }
        }
        __syncthreads();
    }

    if (tid < 16) out[m0 + tid] = llacc[tid];
}

// ---------------------------------------------------------------------------
extern "C" void kernel_launch(void* const* d_in, const int* in_sizes, int n_in,
                              void* d_out, int out_size, void* d_ws, size_t ws_size,
                              hipStream_t stream) {
    (void)in_sizes; (void)n_in; (void)out_size; (void)ws_size;
    const float* x       = (const float*)d_in[0];
    const int*   slen    = (const int*)  d_in[1];
    const float* bmask   = (const float*)d_in[2];
    const float* W_in    = (const float*)d_in[3];
    const float* b_in    = (const float*)d_in[4];
    const float* W_s     = (const float*)d_in[5];
    const float* b_s     = (const float*)d_in[6];
    const float* W_t     = (const float*)d_in[7];
    const float* b_t     = (const float*)d_in[8];
    const float* rescale = (const float*)d_in[9];
    const float* W_res   = (const float*)d_in[10];
    const float* W_inp   = (const float*)d_in[11];
    float*  out = (float*)d_out;
    __bf16* wsb = (__bf16*)d_ws;

    prep_weights<<<(TOTAL_W + 255) / 256, 256, 0, stream>>>(W_in, W_s, W_t, W_res, W_inp, wsb);
    flow_esn_kernel<<<BB / 16, 256, 0, stream>>>(x, slen, bmask, b_in, b_s, b_t,
                                                 rescale, wsb, out);
}
